// MyModel_23227183136795
// MI455X (gfx1250) — matrix operand order repaired and checked by reading, not compiled, not run on hardware
//
#include <hip/hip_runtime.h>
#include <hip/hip_bf16.h>

// ---------------------------------------------------------------------------
// MI455X (gfx1250). All contractions -> v_wmma_f32_16x16x32_f16 (f16 in,
// f32 accumulate). Weights pre-relayouted to WMMA-B-native [N][K] so LDS
// staging is pure 16B-per-lane copies; double-buffered LDS panels staged via
// gfx1250 async global->LDS DMA (register-staged fallback kept).
// ---------------------------------------------------------------------------

typedef _Float16 h16;
typedef __attribute__((ext_vector_type(16))) _Float16 v16h;
typedef __attribute__((ext_vector_type(8)))  _Float16 v8h;
typedef __attribute__((ext_vector_type(8)))  float    v8f;
typedef int v4i __attribute__((vector_size(16)));

#define FLAG_RELU1 1
#define FLAG_RES   2
#define FLAG_RELU2 4
#define BF_SIGMOID 1

#if defined(__gfx1250__) && __has_builtin(__builtin_amdgcn_global_load_async_to_lds_b128)
#define GEMM_ASYNC 1
#else
#define GEMM_ASYNC 0
#endif

#if GEMM_ASYNC
#define GPTR(p) ((__attribute__((address_space(1))) v4i*)(p))
#define LPTR(p) ((__attribute__((address_space(3))) v4i*)(p))
#if __has_builtin(__builtin_amdgcn_s_wait_asynccnt)
#define WAIT_ASYNC() __builtin_amdgcn_s_wait_asynccnt(0)
#else
#define WAIT_ASYNC() asm volatile("s_wait_asynccnt 0" ::: "memory")
#endif
#endif

__device__ __forceinline__ v16h join8(v8h lo, v8h hi) {
  union { v16h v; v8h h[2]; } u;
  u.h[0] = lo; u.h[1] = hi;
  return u.v;
}

// ---------------------------------------------------------------------------
// WMMA GEMM / tap-summed causal|same conv.
//   C[r,n] = epi( sum_{tap,c} A[b, t + tap*dil - shift0, c] * Bt[n, tap*Kc+c] )
// r = (b << wShift) + t,  Bt is [N][taps*Kc] (WMMA-B-native, pre-transposed).
// Workgroup tile 128x64, 8 waves x (32x32), K-step 32, double-buffered LDS,
// async-DMA staging pipelined against WMMA (one barrier per K-step).
// ---------------------------------------------------------------------------
__global__ __launch_bounds__(256, 2)
void k_gemm(const h16* __restrict__ A, int lda, int wShift,
            const h16* __restrict__ Bt, int ldbt,
            const float* __restrict__ bias,
            const h16* __restrict__ res, int ldres,
            h16* __restrict__ C, int ldc,
            int N, int Kc, int taps, int dil, int shift0,
            int flags)
{
  __shared__ __align__(16) h16 lA [2][128 * 40];   // [row][k<32], pad 40
  __shared__ __align__(16) h16 lBT[2][ 64 * 40];   // [n][k<32],  pad 40

  const int tid   = threadIdx.x;
  const int lane  = tid & 31;
  const int wave  = tid >> 5;
  const int waveM = wave >> 1;            // 0..3
  const int waveN = wave & 1;             // 0..1
  const int hs    = lane >> 4;
  const int lr    = lane & 15;
  const int mBlk  = blockIdx.y * 128;
  const int nBlk  = blockIdx.x * 64;
  const int Wlen  = 1 << wShift;

  // per-thread staging geometry: A = two 8-half chunks, B = one 8-half chunk
  const int arow = tid >> 2;              // 0..63 (and +64)
  const int aseg = tid & 3;
  const int r0 = mBlk + arow, r1 = r0 + 64;
  const int at0 = r0 & (Wlen - 1), at1 = r1 & (Wlen - 1);
  const h16* aBase0 = A + (size_t)(r0 - at0) * lda + aseg * 8;
  const h16* aBase1 = A + (size_t)(r1 - at1) * lda + aseg * 8;
  const int bn = tid >> 2, bseg = tid & 3;
  const h16* bRow = Bt + (size_t)(nBlk + bn) * ldbt + bseg * 8;

  const v8h vzero = {};

#if GEMM_ASYNC
  auto stageA = [&](int buf, int tap, int c0) {
    int ts0 = at0 + tap * dil - shift0;
    int ts1 = at1 + tap * dil - shift0;
    h16* d0 = &lA[buf][arow * 40 + aseg * 8];
    h16* d1 = &lA[buf][(arow + 64) * 40 + aseg * 8];
    if (ts0 >= 0 && ts0 < Wlen)
      __builtin_amdgcn_global_load_async_to_lds_b128(
          GPTR(aBase0 + (size_t)ts0 * lda + c0), LPTR(d0), 0, 0);
    else
      *(v8h*)d0 = vzero;
    if (ts1 >= 0 && ts1 < Wlen)
      __builtin_amdgcn_global_load_async_to_lds_b128(
          GPTR(aBase1 + (size_t)ts1 * lda + c0), LPTR(d1), 0, 0);
    else
      *(v8h*)d1 = vzero;
  };
  auto stageB = [&](int buf, int kk) {
    __builtin_amdgcn_global_load_async_to_lds_b128(
        GPTR(bRow + kk), LPTR(&lBT[buf][bn * 40 + bseg * 8]), 0, 0);
  };
#else
  v8h ra0 = vzero, ra1 = vzero, rb = vzero;
  auto loadA = [&](int tap, int c0) {
    int ts0 = at0 + tap * dil - shift0;
    int ts1 = at1 + tap * dil - shift0;
    ra0 = vzero; ra1 = vzero;
    if (ts0 >= 0 && ts0 < Wlen) ra0 = *(const v8h*)(aBase0 + (size_t)ts0 * lda + c0);
    if (ts1 >= 0 && ts1 < Wlen) ra1 = *(const v8h*)(aBase1 + (size_t)ts1 * lda + c0);
  };
  auto storeA = [&](int buf) {
    *(v8h*)&lA[buf][arow * 40 + aseg * 8]        = ra0;
    *(v8h*)&lA[buf][(arow + 64) * 40 + aseg * 8] = ra1;
  };
  auto loadB  = [&](int kk)  { rb = *(const v8h*)(bRow + kk); };
  auto storeB = [&](int buf) { *(v8h*)&lBT[buf][bn * 40 + bseg * 8] = rb; };
#endif

  v8f acc[2][2] = {};
  auto compute = [&](int buf) {
    v16h afr[2], bfr[2];
#pragma unroll
    for (int mt = 0; mt < 2; ++mt) {
      int rr = waveM * 32 + mt * 16 + lr;
      afr[mt] = join8(*(const v8h*)&lA[buf][rr * 40 + hs * 8],
                      *(const v8h*)&lA[buf][rr * 40 + 16 + hs * 8]);
    }
#pragma unroll
    for (int nt = 0; nt < 2; ++nt) {
      int cb = waveN * 32 + nt * 16 + lr;
      bfr[nt] = join8(*(const v8h*)&lBT[buf][cb * 40 + hs * 8],
                      *(const v8h*)&lBT[buf][cb * 40 + 16 + hs * 8]);
    }
#pragma unroll
    for (int mt = 0; mt < 2; ++mt)
#pragma unroll
      for (int nt = 0; nt < 2; ++nt)
        acc[mt][nt] = __builtin_amdgcn_wmma_f32_16x16x32_f16(
            false, afr[mt], false, bfr[nt], (short)0, acc[mt][nt], false, false);
  };

  // ---- software pipeline, one barrier per K-step ----
  int tap = 0, c0 = 0, buf = 0;
#if GEMM_ASYNC
  stageA(0, 0, 0); stageB(0, 0);
  WAIT_ASYNC();
#else
  loadA(0, 0); storeA(0);
  loadB(0);    storeB(0);
#endif
  __syncthreads();
  for (;;) {
    int ntap = tap, nc0 = c0 + 32;
    if (nc0 == Kc) { nc0 = 0; ++ntap; }
    const bool more = (ntap < taps);
    if (more) {
#if GEMM_ASYNC
      stageA(buf ^ 1, ntap, nc0);
      stageB(buf ^ 1, ntap * Kc + nc0);
#else
      loadA(ntap, nc0);
      loadB(ntap * Kc + nc0);
#endif
    }
    compute(buf);
    if (!more) break;
#if GEMM_ASYNC
    WAIT_ASYNC();
#else
    storeA(buf ^ 1);
    storeB(buf ^ 1);
#endif
    __syncthreads();
    buf ^= 1; tap = ntap; c0 = nc0;
  }

  // ---- epilogue: bias, relu, residual, relu ----
#pragma unroll
  for (int mt = 0; mt < 2; ++mt) {
#pragma unroll
    for (int nt = 0; nt < 2; ++nt) {
      int col = nBlk + waveN * 32 + nt * 16 + lr;
      float bv = bias ? bias[col] : 0.0f;
#pragma unroll
      for (int v = 0; v < 8; ++v) {
        int row = mBlk + waveM * 32 + mt * 16 + v + 8 * hs;
        float x = acc[mt][nt][v] + bv;
        if (flags & FLAG_RELU1) x = x > 0.0f ? x : 0.0f;
        if (flags & FLAG_RES)   x += (float)res[(size_t)row * ldres + col];
        if (flags & FLAG_RELU2) x = x > 0.0f ? x : 0.0f;
        C[(size_t)row * ldc + col] = (h16)x;
      }
    }
  }
}

// ---------------------------------------------------------------------------
// Batched small WMMA GEMM, one wave per 16x16 tile. B operand is always in
// WMMA-B-native layout Bt[n][k], so both fragments are two 16B loads.
// Batch offsets: bb -> (bb>>divShift)*S1 + (bb&mask)*S2.
// ---------------------------------------------------------------------------
__global__ __launch_bounds__(32)
void k_bgemm(const h16* __restrict__ A, long aS1, long aS2, int lda,
             const h16* __restrict__ Bt, long bS1, long bS2, int ldb,
             h16* __restrict__ C, long cS1, long cS2, int ldc,
             int Kd, int divShift, float alpha, int flags)
{
  const int bb = blockIdx.z;
  const int hi = bb >> divShift, lo = bb & ((1 << divShift) - 1);
  const long aOff = (long)hi * aS1 + (long)lo * aS2;
  const long bOff = (long)hi * bS1 + (long)lo * bS2;
  const long cOff = (long)hi * cS1 + (long)lo * cS2;
  const int m0 = blockIdx.y * 16, n0 = blockIdx.x * 16;
  const int lane = threadIdx.x & 31, hs = lane >> 4, lr = lane & 15;

  v8f acc = {};
  for (int kk = 0; kk < Kd; kk += 32) {
    const h16* ap = A  + aOff + (size_t)(m0 + lr) * lda + kk;
    const h16* bp = Bt + bOff + (size_t)(n0 + lr) * ldb + kk;
    v16h af = join8(*(const v8h*)(ap + hs * 8),  *(const v8h*)(ap + 16 + hs * 8));
    v16h bf = join8(*(const v8h*)(bp + hs * 8),  *(const v8h*)(bp + 16 + hs * 8));
    acc = __builtin_amdgcn_wmma_f32_16x16x32_f16(
        false, af, false, bf, (short)0, acc, false, false);
  }
#pragma unroll
  for (int v = 0; v < 8; ++v) {
    int row = m0 + v + 8 * hs, col = n0 + lr;
    float x = acc[v] * alpha;
    if (flags & BF_SIGMOID) x = 1.0f / (1.0f + __expf(-x));
    C[cOff + (size_t)row * ldc + col] = (h16)x;
  }
}

// ---------------------------------------------------------------------------
__global__ void k_softmax(h16* __restrict__ p, int L)
{
  const int r = blockIdx.x, t = threadIdx.x;
  __shared__ float red[128];
  h16* row = p + (size_t)r * L;
  float v = (float)row[t];
  red[t] = v; __syncthreads();
  for (int s = 64; s > 0; s >>= 1) { if (t < s) red[t] = fmaxf(red[t], red[t + s]); __syncthreads(); }
  float mx = red[0]; __syncthreads();
  float e = __expf(v - mx);
  red[t] = e; __syncthreads();
  for (int s = 64; s > 0; s >>= 1) { if (t < s) red[t] += red[t + s]; __syncthreads(); }
  row[t] = (h16)(e / red[0]);
}

__global__ void k_layernorm(h16* __restrict__ dst, const h16* __restrict__ a,
                            const h16* __restrict__ b,
                            const float* __restrict__ g, const float* __restrict__ be,
                            int D)
{
  const int r = blockIdx.x, t = threadIdx.x;   // blockDim == D == 256
  __shared__ float red[256];
  float v = (float)a[(size_t)r * D + t];
  if (b) v += (float)b[(size_t)r * D + t];
  red[t] = v; __syncthreads();
  for (int s = 128; s > 0; s >>= 1) { if (t < s) red[t] += red[t + s]; __syncthreads(); }
  float mean = red[0] / (float)D; __syncthreads();
  float d = v - mean;
  red[t] = d * d; __syncthreads();
  for (int s = 128; s > 0; s >>= 1) { if (t < s) red[t] += red[t + s]; __syncthreads(); }
  float var = red[0] / (float)D;
  dst[(size_t)r * D + t] = (h16)(d * rsqrtf(var + 1e-5f) * g[t] + be[t]);
}

__global__ void k_gat_e(const h16* __restrict__ left, const h16* __restrict__ right,
                        const float* __restrict__ gw, h16* __restrict__ e,
                        int F, int Dg)
{
  const int bi = blockIdx.x;           // b*F + i
  const int b  = bi >> 7;              // F == 128
  const int j  = threadIdx.x;
  __shared__ float lrow[256];
  __shared__ float gws[256];
  for (int d = threadIdx.x; d < Dg; d += blockDim.x) {
    lrow[d] = (float)left[(size_t)bi * Dg + d];
    gws[d]  = gw[d];
  }
  __syncthreads();
  const h16* rrow = right + (size_t)(b * F + j) * Dg;
  float acc = 0.0f;
  for (int d = 0; d < Dg; d += 8) {
    v8h rv = *(const v8h*)(rrow + d);
#pragma unroll
    for (int q = 0; q < 8; ++q) {
      float tv = lrow[d + q] + (float)rv[q];
      tv = tv > 0.0f ? tv : 0.5f * tv;
      acc += tv * gws[d + q];
    }
  }
  e[(size_t)bi * F + j] = (h16)acc;
}

__global__ void k_transpose(const h16* __restrict__ src, int srcRS, int srcBS,
                            h16* __restrict__ dst, int dstRS, int dstBS, int dstOff)
{
  __shared__ h16 tile[16][17];
  const int b  = blockIdx.z;
  const int r0 = blockIdx.y * 16, c0 = blockIdx.x * 16;
  const int tx = threadIdx.x, ty = threadIdx.y;
  tile[ty][tx] = src[(size_t)b * srcBS + (size_t)(r0 + ty) * srcRS + c0 + tx];
  __syncthreads();
  dst[(size_t)b * dstBS + (size_t)(c0 + ty) * dstRS + dstOff + r0 + tx] = tile[tx][ty];
}

// f32->f16 straight copy
__global__ void k_cvt(const float* __restrict__ s, h16* __restrict__ d, int n)
{
  int i = blockIdx.x * blockDim.x + threadIdx.x;
  if (i < n) d[i] = (h16)s[i];
}

// f32 [K,N] -> f16 Bt [N,K]
__global__ void k_cvt_t(const float* __restrict__ s, h16* __restrict__ d, int K, int N)
{
  int i = blockIdx.x * blockDim.x + threadIdx.x;
  if (i >= K * N) return;
  int n = i / K, k = i - n * K;
  d[i] = (h16)s[(size_t)k * N + n];
}

// conv weight f32 [O,I,K] -> f16 Bt [o][k*I + i]  (row length = K*I)
__global__ void k_convw_t(const float* __restrict__ s, h16* __restrict__ d,
                          int O, int I, int K)
{
  int i = blockIdx.x * blockDim.x + threadIdx.x;
  if (i >= O * I * K) return;
  int ci = i % I; int t = i / I; int k = t % K; int o = t / K;
  d[i] = (h16)s[(size_t)o * I * K + (size_t)ci * K + k];
}

__global__ void k_final(const h16* __restrict__ y, const h16* __restrict__ w,
                        const float* __restrict__ bias, float* __restrict__ out,
                        int W, int D)
{
  const int b = blockIdx.x, o = threadIdx.x;   // blockDim == D
  const h16* row = y + ((size_t)b * W + (W - 1)) * D;
  float acc = bias[o];
  for (int c = 0; c < D; ++c)
    acc += (float)row[c] * (float)w[(size_t)c * D + o];
  out[b * D + o] = acc;
  if (b == 0 && o == 0) out[16 * D] = 0.0f;
}

// ---------------------------------------------------------------------------
// Host orchestration
// ---------------------------------------------------------------------------
extern "C" void kernel_launch(void* const* d_in, const int* in_sizes, int n_in,
                              void* d_out, int out_size, void* d_ws, size_t ws_size,
                              hipStream_t stream)
{
  (void)in_sizes; (void)n_in; (void)out_size; (void)ws_size;
  enum { B = 16, W = 128, F = 128, D = 256, KT = 5, H = 8, M = B * W };

  const float* x      = (const float*)d_in[0];
  const float* ext_w  = (const float*)d_in[1];
  const float* ext_b  = (const float*)d_in[2];
  const float* gatlw  = (const float*)d_in[3];
  const float* gatlb  = (const float*)d_in[4];
  const float* gatw   = (const float*)d_in[5];
  const float* tinw   = (const float*)d_in[6];
  const float* tinb   = (const float*)d_in[7];
  const float* wq = (const float*)d_in[8];  const float* bq = (const float*)d_in[9];
  const float* wk = (const float*)d_in[10]; const float* bk = (const float*)d_in[11];
  const float* wv = (const float*)d_in[12]; const float* bv = (const float*)d_in[13];
  const float* wo = (const float*)d_in[14]; const float* bo = (const float*)d_in[15];
  const float* w1 = (const float*)d_in[16]; const float* b1 = (const float*)d_in[17];
  const float* w2 = (const float*)d_in[18]; const float* b2 = (const float*)d_in[19];
  const float* g1 = (const float*)d_in[20]; const float* e1 = (const float*)d_in[21];
  const float* g2 = (const float*)d_in[22]; const float* e2 = (const float*)d_in[23];
  const float* fng = (const float*)d_in[24]; const float* fnb = (const float*)d_in[25];
  const float* outw = (const float*)d_in[26]; const float* outb = (const float*)d_in[27];
  const float *c1w[4], *c1b[4], *c2w[4], *c2b[4], *dsw = nullptr, *dsb = nullptr;
  int idx = 28;
  for (int blk = 0; blk < 4; ++blk) {
    c1w[blk] = (const float*)d_in[idx++]; c1b[blk] = (const float*)d_in[idx++];
    c2w[blk] = (const float*)d_in[idx++]; c2b[blk] = (const float*)d_in[idx++];
    if (blk == 0) { dsw = (const float*)d_in[idx++]; dsb = (const float*)d_in[idx++]; }
  }
  const float* linw = (const float*)d_in[idx++];
  const float* linb = (const float*)d_in[idx++];

  char* base = (char*)d_ws;
  size_t off = 0;
  auto alloc = [&](size_t halves) -> h16* {
    h16* p = (h16*)(base + off);
    off += ((halves * sizeof(h16) + 255) & ~(size_t)255);
    return p;
  };

  h16* xh    = alloc((size_t)B * W * F);
  h16* wexth = alloc((size_t)F * F * KT);        // [128][640]
  h16* wgat1 = alloc(256 * 128);                 // A1^T [256][128]
  h16* wgat2 = alloc(256 * 128);                 // A2^T [256][128]
  h16* wtinh = alloc(D * F);                     // [256][128]
  h16* wqh = alloc(2 * D * D); h16* wkh = alloc(2 * D * D);
  h16* wvh = alloc(2 * D * D); h16* woh = alloc(2 * D * D);
  h16* w1h = alloc(2 * 2 * D * D);               // per layer [512][256]
  h16* w2h = alloc(2 * 2 * D * D);               // per layer [256][512]
  h16* wouth = alloc(F * D);                     // [128][256]
  h16* wcv1[4]; h16* wcv2[4];
  wcv1[0] = alloc((size_t)256 * 384 * KT);       // [256][1920]
  wcv2[0] = alloc((size_t)256 * 256 * KT);       // [256][1280]
  h16* wdsh = alloc(256 * 384);                  // [256][384]
  for (int blk = 1; blk < 4; ++blk) {
    wcv1[blk] = alloc((size_t)256 * 256 * KT);
    wcv2[blk] = alloc((size_t)256 * 256 * KT);
  }
  h16* wlinh = alloc(D * D);                     // plain [c][o] for k_final

  h16* ycat = alloc((size_t)B * W * 384);
  h16* h0 = alloc((size_t)M * D); h16* h1 = alloc((size_t)M * D);
  h16* qb = alloc((size_t)M * D); h16* kb = alloc((size_t)M * D); h16* vb = alloc((size_t)M * D);
  h16* vT = alloc((size_t)M * D);                // per-batch transposed V [B][D][W]
  h16* sc = alloc((size_t)B * H * W * W);
  h16* ob = alloc((size_t)M * D); h16* ob2 = alloc((size_t)M * D);
  h16* f1 = alloc((size_t)M * 2 * D); h16* f2 = alloc((size_t)M * D);
  h16* leftb = alloc((size_t)M * 256); h16* rightb = alloc((size_t)M * 256);
  h16* eG  = alloc((size_t)B * F * F);
  h16* hfT = alloc((size_t)B * F * W);
  h16* vt  = alloc((size_t)B * F * W);
  h16* t0 = alloc((size_t)M * D); h16* t1 = alloc((size_t)M * D);
  h16* t2 = alloc((size_t)M * D); h16* res0 = alloc((size_t)M * D);

  auto cvt = [&](const float* s, h16* d, int n) {
    k_cvt<<<dim3((n + 255) / 256), dim3(256), 0, stream>>>(s, d, n);
  };
  auto cvtT = [&](const float* s, h16* d, int K, int N) {
    int n = K * N;
    k_cvt_t<<<dim3((n + 255) / 256), dim3(256), 0, stream>>>(s, d, K, N);
  };
  auto convwT = [&](const float* s, h16* d, int O, int I, int Kk) {
    int n = O * I * Kk;
    k_convw_t<<<dim3((n + 255) / 256), dim3(256), 0, stream>>>(s, d, O, I, Kk);
  };
  auto gemm = [&](const h16* A, int lda, int wShift, const h16* Bt, int ldbt,
                  const float* bias, const h16* res, int ldres, h16* C, int ldc,
                  int Mm, int Nn, int Kc, int taps, int dil, int shift0, int flags) {
    k_gemm<<<dim3(Nn / 64, Mm / 128), dim3(256), 0, stream>>>(
        A, lda, wShift, Bt, ldbt, bias, res, ldres, C, ldc,
        Nn, Kc, taps, dil, shift0, flags);
  };

  // ---- conversions: all weights to f16, WMMA-B-native [N][K] layout ----
  cvt(x, xh, B * W * F);
  convwT(ext_w, wexth, F, F, KT);
  cvtT(gatlw,             wgat1, 128, 256);
  cvtT(gatlw + 128 * 256, wgat2, 128, 256);
  cvtT(tinw, wtinh, 128, 256);
  for (int l = 0; l < 2; ++l) {
    cvtT(wq + (size_t)l * D * D, wqh + (size_t)l * D * D, D, D);
    cvtT(wk + (size_t)l * D * D, wkh + (size_t)l * D * D, D, D);
    cvtT(wv + (size_t)l * D * D, wvh + (size_t)l * D * D, D, D);
    cvtT(wo + (size_t)l * D * D, woh + (size_t)l * D * D, D, D);
    cvtT(w1 + (size_t)l * 2 * D * D, w1h + (size_t)l * 2 * D * D, 256, 512);
    cvtT(w2 + (size_t)l * 2 * D * D, w2h + (size_t)l * 2 * D * D, 512, 256);
  }
  cvtT(outw, wouth, 256, 128);
  convwT(c1w[0], wcv1[0], 256, 384, KT);
  convwT(c2w[0], wcv2[0], 256, 256, KT);
  convwT(dsw, wdsh, 256, 384, 1);
  for (int blk = 1; blk < 4; ++blk) {
    convwT(c1w[blk], wcv1[blk], 256, 256, KT);
    convwT(c2w[blk], wcv2[blk], 256, 256, KT);
  }
  cvt(linw, wlinh, D * D);

  // ---- feature extractor: K=5 same-pad conv as tap-summed WMMA GEMM ----
  gemm(xh, F, 7, wexth, 640, ext_b, nullptr, 0, ycat, 384,
       M, F, F, KT, 1, 2, FLAG_RELU1);           // v -> ycat[:,:,0:128]

  // ---- transformer ----
  gemm(ycat, 384, 11, wtinh, 128, tinb, nullptr, 0, h0, D, M, D, F, 1, 0, 0, 0);
  for (int l = 0; l < 2; ++l) {
    gemm(h0, D, 11, wqh + (size_t)l * D * D, D, bq + l * D, nullptr, 0, qb, D, M, D, D, 1, 0, 0, 0);
    gemm(h0, D, 11, wkh + (size_t)l * D * D, D, bk + l * D, nullptr, 0, kb, D, M, D, D, 1, 0, 0, 0);
    gemm(h0, D, 11, wvh + (size_t)l * D * D, D, bv + l * D, nullptr, 0, vb, D, M, D, D, 1, 0, 0, 0);
    // scores: per (b,h) Q[128,32] @ K^T  (K rows are already B-native)
    k_bgemm<<<dim3(W / 16, W / 16, B * H), dim3(32), 0, stream>>>(
        qb, (long)W * D, 32L, D,
        kb, (long)W * D, 32L, D,
        sc, (long)H * W * W, (long)W * W, W,
        32, 3, 0.17677669529663687f, 0);
    k_softmax<<<dim3(B * H * W), dim3(W), 0, stream>>>(sc, W);
    // V^T per batch: vT[b][d][j] = V[b][j][d]
    k_transpose<<<dim3(D / 16, W / 16, B), dim3(16, 16), 0, stream>>>(
        vb, D, W * D, vT, W, D * W, 0);
    // O = attn @ V : per (b,h) [128,128] @ [128,32], B-native from vT
    k_bgemm<<<dim3(32 / 16, W / 16, B * H), dim3(32), 0, stream>>>(
        sc, (long)H * W * W, (long)W * W, W,
        vT, (long)D * W, 32L * W, W,
        ob, (long)W * D, 32L, D,
        W, 3, 1.0f, 0);
    gemm(ob, D, 11, woh + (size_t)l * D * D, D, bo + l * D, nullptr, 0, ob2, D, M, D, D, 1, 0, 0, 0);
    k_layernorm<<<dim3(M), dim3(D), 0, stream>>>(h1, h0, ob2, g1 + l * D, e1 + l * D, D);
    gemm(h1, D, 11, w1h + (size_t)l * 2 * D * D, D, b1 + l * 2 * D, nullptr, 0,
         f1, 2 * D, M, 2 * D, D, 1, 0, 0, FLAG_RELU1);
    gemm(f1, 2 * D, 11, w2h + (size_t)l * 2 * D * D, 2 * D, b2 + l * D, nullptr, 0,
         f2, D, M, D, 2 * D, 1, 0, 0, 0);
    k_layernorm<<<dim3(M), dim3(D), 0, stream>>>(h0, h1, f2, g2 + l * D, e2 + l * D, D);
  }
  k_layernorm<<<dim3(M), dim3(D), 0, stream>>>(h1, h0, nullptr, fng, fnb, D);
  gemm(h1, D, 11, wouth, D, outb, nullptr, 0, ycat + 128, 384,
       M, F, D, 1, 0, 0, 0);                     // h_temp -> ycat[:,:,128:256]

  // ---- GAT ----
  k_transpose<<<dim3(F / 16, W / 16, B), dim3(16, 16), 0, stream>>>(
      ycat, 384, W * 384, vt, W, F * W, 0);      // vt[b][f][w] = v[b][w][f]
  gemm(vt, W, 11, wgat1, W, gatlb, nullptr, 0, leftb, 256, M, 256, W, 1, 0, 0, 0);
  gemm(vt, W, 11, wgat2, W, nullptr, nullptr, 0, rightb, 256, M, 256, W, 1, 0, 0, 0);
  k_gat_e<<<dim3(B * F), dim3(F), 0, stream>>>(leftb, rightb, gatw, eG, F, 256);
  k_softmax<<<dim3(B * F), dim3(F), 0, stream>>>(eG, F);
  // h_feat^T[b] = sigmoid(attn[b] @ vt[b]); B-native operand is v == ycat[:,:,0:128]
  k_bgemm<<<dim3(W / 16, F / 16, B), dim3(32), 0, stream>>>(
      eG, (long)F * F, 0L, F,
      ycat, (long)W * 384, 0L, 384,
      hfT, (long)F * W, 0L, W,
      F, 0, 1.0f, BF_SIGMOID);
  k_transpose<<<dim3(W / 16, F / 16, B), dim3(16, 16), 0, stream>>>(
      hfT, W, F * W, ycat, 384, W * 384, 256);   // -> ycat[:,:,256:384]

  // ---- TCN ----
  gemm(ycat, 384, 7, wdsh, 384, dsb, nullptr, 0, res0, 256, M, 256, 384, 1, 0, 0, 0);
  gemm(ycat, 384, 7, wcv1[0], 1920, c1b[0], nullptr, 0, t0, 256,
       M, 256, 384, KT, 1, 4, FLAG_RELU1);
  gemm(t0, 256, 7, wcv2[0], 1280, c2b[0], res0, 256, t1, 256,
       M, 256, 256, KT, 1, 4, FLAG_RELU1 | FLAG_RES | FLAG_RELU2);
  h16* yin = t1; h16* pong = t2;
  for (int blk = 1; blk < 4; ++blk) {
    int dil = 1 << blk;
    gemm(yin, 256, 7, wcv1[blk], 1280, c1b[blk], nullptr, 0, t0, 256,
         M, 256, 256, KT, dil, 4 * dil, FLAG_RELU1);
    gemm(t0, 256, 7, wcv2[blk], 1280, c2b[blk], yin, 256, pong, 256,
         M, 256, 256, KT, dil, 4 * dil, FLAG_RELU1 | FLAG_RES | FLAG_RELU2);
    h16* tmp = yin; yin = pong; pong = tmp;
  }

  k_final<<<dim3(B), dim3(D), 0, stream>>>(yin, wlinh, linb, (float*)d_out, W, D);
}
